// KAN_Convolutional_Layer_11321533792345
// MI455X (gfx1250) — matrix operand order repaired and checked by reading, not compiled, not run on hardware
//
#include <hip/hip_runtime.h>
#include <hip/hip_bf16.h>

// ---------------------------------------------------------------------------
// KAN conv layer for MI455X (gfx1250, wave32, WMMA).
// Feature vector per point (reordered, 16B-aligned basis blocks):
//   [bspline basis k=0..8 -> slots 8k..8k+7 (72)] ++ [silu(p_k) -> 72+k (9)]
// padded to K=96 (pad weights are zero -> pad garbage contributes 0).
// Contraction vs 8 convs as 16x16 WMMA tiles:
//   D(16 pts x 16 ch) = A(16 x 96 f16) * B(96 x 16 f16), 3x K=32 chunks.
// Uniform grid => closed-form cubic B-spline blending (4 nonzeros/scalar).
// ---------------------------------------------------------------------------

typedef __attribute__((ext_vector_type(16))) _Float16 v16h;
typedef __attribute__((ext_vector_type(8)))  float    v8f;

#define B_      16
#define C_      32
#define H_      64
#define W_      64
#define KER     3
#define KK      9
#define NCONV   8
#define SCOEF   8
#define HO      62
#define WO      62
#define LL      (HO * WO)          // 3844
#define NPTS    (B_ * C_ * LL)     // 1,968,128
#define NFEAT   81
#define KPAD    96
#define WAVES   8
#define TILES   (NPTS / 16)        // 123,008
#define BLOCKS  (TILES / WAVES)    // 15,376

union FragH { v16h v; float4 q[2]; };

__global__ __launch_bounds__(256) void kan_conv_wmma_kernel(
    const float* __restrict__ x,
    const float* __restrict__ base_w,        // (8,9)
    const float* __restrict__ spline_w,      // (8,9,8)
    const float* __restrict__ spline_scaler, // (8,9)
    float* __restrict__ out)                 // (16, 256, 62, 62)
{
    __shared__ _Float16 sW[16 * KPAD];           // packed weights Wt[n][k]
    __shared__ _Float16 sFeat[WAVES][16 * KPAD]; // per-wave feature tiles
    __shared__ float    sOut[WAVES][16 * 16];    // per-wave D staging

    const int tid  = threadIdx.x;
    const int wave = tid >> 5;
    const int lane = tid & 31;

    // ---- Build packed weight matrix Wt[n][k] (n<8 real, rest zero) --------
    for (int idx = tid; idx < 16 * KPAD; idx += 256) {
        const int n = idx / KPAD;
        const int k = idx % KPAD;
        float v = 0.0f;
        if (n < NCONV) {
            if (k < 72) {               // spline part: basis slot (kk, s)
                const int kk = k >> 3;
                const int s  = k & 7;
                v = spline_w[(n * KK + kk) * SCOEF + s] * spline_scaler[n * KK + kk];
            } else if (k < NFEAT) {     // base (silu) part
                v = base_w[n * KK + (k - 72)];
            }
        }
        sW[idx] = (_Float16)v;
    }
    __syncthreads();

    // ---- Per-wave tile of 16 points ---------------------------------------
    const int tileId = blockIdx.x * WAVES + wave;
    const int base   = tileId * 16;
    const int m      = lane & 15;
    const int pIdx   = base + m;

    const int l  = pIdx % LL;
    const int bc = pIdx / LL;                 // b*C + c
    const int oh = l / WO;
    const int ow = l % WO;
    const float* xrow = x + (bc * H_ + oh) * W_ + ow;

    _Float16* feat = &sFeat[wave][m * KPAD];

    // Uniform 5-iteration split: lanes<16 do k=0..4, lanes>=16 do k=4..8.
    // k=4 overlap is lockstep-identical writes (benign).
    const int kbase = (lane < 16) ? 0 : 4;
    const float4 zero4 = make_float4(0.f, 0.f, 0.f, 0.f);

#pragma unroll
    for (int j = 0; j < 5; ++j) {
        const int k  = kbase + j;
        const int kh = k / KER;
        const int kw = k % KER;
        const float xv = xrow[kh * W_ + kw];

        // SiLU -> slot 72+k
        const float silu = xv / (1.0f + __expf(-xv));
        feat[72 + k] = (_Float16)silu;

        // Zero this k's 8 basis slots (16B aligned)
        *(float4*)(feat + k * 8) = zero4;

        // Closed-form uniform cubic B-spline: t = (x - g0)/h, g0=-2.2, h=0.4
        const float t = (xv + 2.2f) * 2.5f;
        const bool inR = (t >= 0.0f) && (t < 11.0f);
        const float tc = fminf(fmaxf(t, 0.0f), 10.99f);
        const int   i  = (int)tc;              // interval 0..10
        const float u  = tc - (float)i;
        const float u2 = u * u;
        const float u3 = u2 * u;
        const float c6 = 1.0f / 6.0f;
        float w0 = c6 * (1.0f - 3.0f * u + 3.0f * u2 - u3);   // basis i-3
        float w1 = c6 * (4.0f - 6.0f * u2 + 3.0f * u3);       // basis i-2
        float w2 = c6 * (1.0f + 3.0f * u + 3.0f * u2 - 3.0f * u3); // basis i-1
        float w3 = c6 * u3;                                   // basis i
        if (!inR) { w0 = w1 = w2 = w3 = 0.0f; }

        // Scatter 4 nonzeros; out-of-range basis index -> pad dump (weights=0)
        const int s0 = i - 3;
#pragma unroll
        for (int jj = 0; jj < 4; ++jj) {
            const int s   = s0 + jj;
            const float w = (jj == 0) ? w0 : (jj == 1) ? w1 : (jj == 2) ? w2 : w3;
            const int idx = ((unsigned)s <= 7u) ? (k * 8 + s) : (88 + jj);
            feat[idx] = (_Float16)w;
        }
    }
    // zero-pad K = 81..95 (split between the two lanes of this point);
    // pad dump slots may later hold finite garbage -> multiplied by zero weight.
    {
        const int z0 = (lane < 16) ? NFEAT : 88;
        const int z1 = (lane < 16) ? 88 : KPAD;
        for (int z = z0; z < z1; ++z) feat[z] = (_Float16)0.0f;
    }
    __syncthreads();

    // ---- WMMA: D(16x16) = A(16x96) * B(96x16), 3 chained K=32 chunks ------
    v8f acc = {};
    const _Float16* fRow = &sFeat[wave][m * KPAD];
    const _Float16* wRow = &sW[(lane & 15) * KPAD];
#pragma unroll
    for (int ch = 0; ch < 3; ++ch) {
        FragH a, b;
        // A (16-bit 16x32): lanes 0-15 K=c*32+[0..7],[16..23]; lanes 16-31 +8
        const int offA = ch * 32 + ((lane < 16) ? 0 : 8);
        a.q[0] = *(const float4*)(fRow + offA);
        a.q[1] = *(const float4*)(fRow + offA + 16);
        // B (16-bit 32x16): lanes 0-15 col N=lane K=c*32+[0..7],[16..23]; lanes 16-31 +8
        const int offB = ch * 32 + ((lane < 16) ? 0 : 8);
        b.q[0] = *(const float4*)(wRow + offB);
        b.q[1] = *(const float4*)(wRow + offB + 16);
        acc = __builtin_amdgcn_wmma_f32_16x16x32_f16(
            false, a.v, false, b.v, (short)0, acc, false, false);
    }

    // ---- Stage D through LDS for coalesced stores -------------------------
    {
        float* ot = sOut[wave];
        const int n  = lane & 15;
        const int mh = (lane < 16) ? 0 : 8;
#pragma unroll
        for (int r = 0; r < 8; ++r)
            ot[(mh + r) * 16 + n] = acc[r];
    }
    __syncthreads();

    // 128 useful outputs per tile; 4 rounds of 32 lanes; 16 consecutive l per
    // channel per round -> 64B coalesced runs.
    {
        const float* ot = sOut[wave];
#pragma unroll
        for (int it = 0; it < 4; ++it) {
            const int idx = it * 32 + lane;
            const int n   = idx >> 4;
            const int mm  = idx & 15;
            const int p2  = base + mm;
            const int l2  = p2 % LL;
            const int bc2 = p2 / LL;
            out[(bc2 * NCONV + n) * LL + l2] = ot[mm * 16 + n];
        }
    }
}

extern "C" void kernel_launch(void* const* d_in, const int* in_sizes, int n_in,
                              void* d_out, int out_size, void* d_ws, size_t ws_size,
                              hipStream_t stream) {
    (void)in_sizes; (void)n_in; (void)d_ws; (void)ws_size; (void)out_size;
    const float* x  = (const float*)d_in[0];
    const float* bw = (const float*)d_in[1];
    const float* sw = (const float*)d_in[2];
    const float* ss = (const float*)d_in[3];
    float* out = (float*)d_out;
    kan_conv_wmma_kernel<<<BLOCKS, 256, 0, stream>>>(x, bw, sw, ss, out);
}
